// Attention_2860448219393
// MI455X (gfx1250) — matrix operand order repaired and checked by reading, not compiled, not run on hardware
//
#include <hip/hip_runtime.h>

// MI455X (gfx1250) flash attention, wave32 WMMA f16->f32 + TDM staging.
// Kernel 1: Q = X Wq^T + bq (row-major f16), K likewise, V^T (f16, [head][64][2048]).
// Kernel 2: per 16-query tile per wave, transposed flash attention:
//   - K / V^T key chunks staged into LDS once per workgroup by the Tensor Data
//     Mover (tensor_load_to_lds, TENSORcnt), double-buffered.
//   S^T = K Q^T  (A = K rows from LDS, B = Q^T columns from global)
//   online softmax over keys (rows of S^T: in-lane reduce + one xor16 shuffle)
//   O^T += V^T P^T (A = V^T rows from LDS, B = P^T built via xor16 shuffles)
//   out = O^T / (l * 8)     // softmax-then-divide-by-temperature, per reference

typedef _Float16 f16;
typedef __attribute__((ext_vector_type(16))) _Float16 v16h;
typedef __attribute__((ext_vector_type(8)))  _Float16 v8h;
typedef __attribute__((ext_vector_type(8)))  float    v8f;
typedef __attribute__((ext_vector_type(4)))  float    v4f;
typedef __attribute__((ext_vector_type(4)))  unsigned int u32x4;
typedef __attribute__((ext_vector_type(8)))  int      i32x8;
typedef __attribute__((ext_vector_type(4)))  int      i32x4;

#define WMMA_F16(a, b, c) \
  __builtin_amdgcn_wmma_f32_16x16x32_f16(false, (a), false, (b), (short)0, (c), false, false)

#define HEADS_TOTAL 64      // 4 batches * 16 heads
#define SEQ 2048
#define HD  64
#define HEAD_ELEMS (SEQ * HD)   // 131072
#define KCHUNK 64               // keys staged per TDM round
// LDS layout (halfs): buffer b at b*8192: [K tile 64x64 = 4096][VT tile 64x64 = 4096]
#define LDS_BUF_HALFS 8192

// ---------------------------------------------------------------------------
// Kernel 1: QKV projection. One wave = 16 rows of X (16x64 tile).
// ---------------------------------------------------------------------------
__global__ __launch_bounds__(256) void qkv_proj_kernel(
    const float* __restrict__ x,
    const float* __restrict__ Wq, const float* __restrict__ bq,
    const float* __restrict__ Wk, const float* __restrict__ bk,
    const float* __restrict__ Wv, const float* __restrict__ bv,
    f16* __restrict__ Qh, f16* __restrict__ Kh, f16* __restrict__ VTh)
{
  const int lane = threadIdx.x & 31;
  const int wv   = threadIdx.x >> 5;
  const int hi   = (lane >> 4) & 1;
  const int l15  = lane & 15;
  const long tile  = (long)blockIdx.x * 8 + wv;   // 16-row tile index
  const long nbase = tile * 16;
  const int  g     = (int)(nbase >> 11);          // head block (2048 rows each)
  const int  nloc  = (int)(nbase & 2047);
  const long row   = nbase + l15;                 // this lane's X row

  // X tile as B-fragments (X^T: d x n). Lane: col n = row, d = df*32 + 8*hi + i (i<8), df*32 + 16 + 8*hi + (i-8) (i>=8).
  v16h Bx[2];
#pragma unroll
  for (int df = 0; df < 2; ++df) {
    const float* p = x + row * HD + df * 32 + hi * 8;
    v16h t;
#pragma unroll
    for (int i = 0; i < 8; ++i) { t[i] = (f16)p[i]; t[8 + i] = (f16)p[16 + i]; }
    Bx[df] = t;
  }
  // X tile as A-fragments, derived from B-fragments with xor16 lane shuffles.
  v16h Ax[2];
#pragma unroll
  for (int df = 0; df < 2; ++df) {
    union { v16h h; unsigned int u[8]; } src, dst;
    unsigned int sw[8];
    src.h = Bx[df];
#pragma unroll
    for (int j = 0; j < 8; ++j) sw[j] = __shfl_xor(src.u[j], 16, 32);
#pragma unroll
    for (int j = 0; j < 4; ++j) {
      dst.u[j]     = hi ? sw[j + 4] : src.u[j];       // lo: d0..7   hi: d16..23
      dst.u[j + 4] = hi ? src.u[j + 4] : sw[j];       // lo: d8..15 hi: d24..31
    }
    Ax[df] = dst.h;
  }

  // ---- Q and K: Y^T = W @ X^T (rows e, cols n); store row-major [n][64] ----
  const float* Wm[2] = {Wq, Wk};
  const float* bm[2] = {bq, bk};
  f16*         Om[2] = {Qh, Kh};
#pragma unroll
  for (int m = 0; m < 2; ++m) {
#pragma unroll
    for (int eb = 0; eb < 4; ++eb) {
      v16h aW[2];
#pragma unroll
      for (int df = 0; df < 2; ++df) {
        const float* p = Wm[m] + (eb * 16 + l15) * HD + df * 32 + hi * 8;
        v16h t;
#pragma unroll
        for (int i = 0; i < 8; ++i) t[i] = (f16)p[i];
#pragma unroll
        for (int i = 0; i < 8; ++i) t[8 + i] = (f16)p[16 + i];
        aW[df] = t;
      }
      v8f c = {};
      c = WMMA_F16(aW[0], Bx[0], c);
      c = WMMA_F16(aW[1], Bx[1], c);
      const float* bp = bm[m] + eb * 16 + hi * 8;   // rows e = eb*16 + 8*hi + r
      v8h oh;
#pragma unroll
      for (int r = 0; r < 8; ++r) oh[r] = (f16)(c[r] + bp[r]);
      // col n = row fixed per lane; 8 consecutive e values -> packed 16B store
      *(v8h*)(Om[m] + row * HD + eb * 16 + hi * 8) = oh;
    }
  }

  // ---- V: Y = X @ Wv^T (rows n, cols e); store transposed VT[g][e][n] ----
#pragma unroll
  for (int eb = 0; eb < 4; ++eb) {
    v16h bW[2];
#pragma unroll
    for (int df = 0; df < 2; ++df) {
      // B = Wv^T: col e = eb*16+l15, contraction d = df*32 + 16*hi + i
      const float* p = Wv + (eb * 16 + l15) * HD + df * 32 + hi * 16;
      v16h t;
#pragma unroll
      for (int i = 0; i < 16; ++i) t[i] = (f16)p[i];
      bW[df] = t;
    }
    v8f c = {};
    c = WMMA_F16(Ax[0], bW[0], c);
    c = WMMA_F16(Ax[1], bW[1], c);
    const float bias = bv[eb * 16 + l15];           // col e fixed per lane
    v8h oh;
#pragma unroll
    for (int r = 0; r < 8; ++r) oh[r] = (f16)(c[r] + bias);
    // rows n = nloc + 8*hi + r consecutive -> packed 16B store into VT[e][n]
    *(v8h*)(VTh + (long)g * HEAD_ELEMS + (eb * 16 + l15) * SEQ + nloc + hi * 8) = oh;
  }
}

// ---------------------------------------------------------------------------
// TDM: 2-D tile load global -> LDS (data_size = 2 bytes). Dims/strides are in
// elements, addresses in bytes. D# packing per CDNA5 ISA.
// This toolchain exposes the 6-arg builtin:
//   (uint32x4 g0, int32x8 g1, int32x4 g2, int32x4 g3, int32x8 pad, i32 cpol)
// ---------------------------------------------------------------------------
__device__ __forceinline__ void tdm_load_2d_f16(
    unsigned lds_byte_off, unsigned long long gaddr,
    unsigned tensor_d0, unsigned tensor_d1,
    unsigned tile_d0, unsigned tile_d1,
    unsigned long long stride0_elems)
{
  u32x4 g0;
  g0[0] = 1u;                                          // count=1, user mode
  g0[1] = lds_byte_off;                                // lds_addr (bytes)
  g0[2] = (unsigned)(gaddr & 0xffffffffull);           // global_addr[31:0]
  g0[3] = (unsigned)((gaddr >> 32) & 0x1ffffffull)     // global_addr[56:32]
        | (2u << 30);                                  // type = 2 ("image")
  i32x8 g1;
  g1[0] = (int)(1u << 16);                             // wg_mask=0, data_size=1 (2B)
  g1[1] = (int)((tensor_d0 & 0xffffu) << 16);          // tensor_dim0[15:0]
  g1[2] = (int)(((tensor_d0 >> 16) & 0xffffu) | ((tensor_d1 & 0xffffu) << 16));
  g1[3] = (int)(((tensor_d1 >> 16) & 0xffffu) | ((tile_d0 & 0xffffu) << 16));
  g1[4] = (int)(tile_d1 & 0xffffu);                    // tile_dim1; tile_dim2=0
  g1[5] = (int)(unsigned)(stride0_elems & 0xffffffffull);
  g1[6] = (int)(unsigned)((stride0_elems >> 32) & 0xffffull);  // stride1=0
  g1[7] = 0;
  i32x4 gz = {0, 0, 0, 0};                             // groups 2/3 unused (2-D)
  i32x8 gp = {0, 0, 0, 0, 0, 0, 0, 0};
  __builtin_amdgcn_tensor_load_to_lds(g0, g1, gz, gz, gp, 0);
}

// ---------------------------------------------------------------------------
// Kernel 2: transposed flash attention with TDM-staged K / V^T tiles.
// Block = 128 queries of one head (8 waves x 16 queries).
// ---------------------------------------------------------------------------
__global__ __launch_bounds__(256) void attn_kernel(
    const f16* __restrict__ Qh, const f16* __restrict__ Kh,
    const f16* __restrict__ VTh, float* __restrict__ out)
{
  extern __shared__ f16 smem[];   // 2 * LDS_BUF_HALFS halfs (32 KB), offset 0

  const int lane = threadIdx.x & 31;
  const int wv   = threadIdx.x >> 5;
  const int hi   = (lane >> 4) & 1;
  const int l15  = lane & 15;
  const int g     = blockIdx.x >> 4;     // head index 0..63
  const int qbase = (blockIdx.x & 15) * 128 + wv * 16;
  const long go   = (long)g * HEAD_ELEMS;

  // Q as B-fragments of Q^T (contraction d, cols q). One 32B load per frag.
  const f16* qp = Qh + go + (long)(qbase + l15) * HD;
  v16h bQ0 = *(const v16h*)(qp + hi * 16);
  v16h bQ1 = *(const v16h*)(qp + 32 + hi * 16);

  v8f o0 = {}, o1 = {}, o2 = {}, o3 = {};
  float m = -1e30f, l = 0.0f;

  // Prologue: stage chunk 0 into buffer 0 (wave 0 drives the TDM).
  if (wv == 0) {
    // K rows kb..kb+63, 64 halfs each, contiguous (stride 64 elements)
    tdm_load_2d_f16(0u, (unsigned long long)(size_t)(Kh + go),
                    HD, SEQ, HD, KCHUNK, HD);
    // VT rows d=0..63, 64 halfs starting at col 0, row stride 2048 elements
    tdm_load_2d_f16(4096u * 2u, (unsigned long long)(size_t)(VTh + go),
                    SEQ, HD, KCHUNK, HD, SEQ);
    __builtin_amdgcn_s_wait_tensorcnt(0);
  }
  __syncthreads();

  for (int kb2 = 0; kb2 < SEQ; kb2 += KCHUNK) {
    const int cur = (kb2 >> 6) & 1;
    const bool more = (kb2 + KCHUNK) < SEQ;
    if (wv == 0 && more) {
      const unsigned nb = (unsigned)(cur ^ 1) * (LDS_BUF_HALFS * 2u); // bytes
      const long nkb = kb2 + KCHUNK;
      tdm_load_2d_f16(nb, (unsigned long long)(size_t)(Kh + go + nkb * HD),
                      HD, SEQ, HD, KCHUNK, HD);
      tdm_load_2d_f16(nb + 4096u * 2u,
                      (unsigned long long)(size_t)(VTh + go + nkb),
                      SEQ, HD, KCHUNK, HD, SEQ);
    }
    const f16* Kt = smem + cur * LDS_BUF_HALFS;          // [64][64]
    const f16* Vt = Kt + 4096;                           // [64][64]

#pragma unroll
    for (int ks = 0; ks < KCHUNK; ks += 32) {
      // S^T = K Q^T : two 16x16 C frags (key groups ks..+15, ks+16..+31)
      v8f st0, st1;
      {
        union { v16h h; v8h q[2]; } a;
        v8f c = {};
        const f16* p = Kt + (ks + l15) * HD + hi * 16;
        a.q[0] = *(const v8h*)(p);       a.q[1] = *(const v8h*)(p + 8);
        c = WMMA_F16(a.h, bQ0, c);
        a.q[0] = *(const v8h*)(p + 32);  a.q[1] = *(const v8h*)(p + 40);
        c = WMMA_F16(a.h, bQ1, c);
        st0 = c;
        v8f c1 = {};
        const f16* p1 = Kt + (ks + 16 + l15) * HD + hi * 16;
        a.q[0] = *(const v8h*)(p1);      a.q[1] = *(const v8h*)(p1 + 8);
        c1 = WMMA_F16(a.h, bQ0, c1);
        a.q[0] = *(const v8h*)(p1 + 32); a.q[1] = *(const v8h*)(p1 + 40);
        c1 = WMMA_F16(a.h, bQ1, c1);
        st1 = c1;
      }

      // Online softmax over keys (rows of S^T): in-lane + one xor16 combine.
      float mloc = -1e30f;
#pragma unroll
      for (int r = 0; r < 8; ++r) {
        mloc = fmaxf(mloc, st0[r]);
        mloc = fmaxf(mloc, st1[r]);
      }
      mloc = fmaxf(mloc, __shfl_xor(mloc, 16, 32));
      const float mnew = fmaxf(m, mloc);
      const float esc  = __expf(m - mnew);
      float p0[8], p1[8], s = 0.0f;
#pragma unroll
      for (int r = 0; r < 8; ++r) {
        p0[r] = __expf(st0[r] - mnew);
        p1[r] = __expf(st1[r] - mnew);
        s += p0[r] + p1[r];
      }
      s += __shfl_xor(s, 16, 32);
      l = l * esc + s;
      m = mnew;
#pragma unroll
      for (int r = 0; r < 8; ++r) {
        o0[r] *= esc; o1[r] *= esc; o2[r] *= esc; o3[r] *= esc;
      }

      // Build B-fragment of P^T (32k x 16q) from exp'd S^T via xor16 shuffles.
      float p0s[8], p1s[8];
#pragma unroll
      for (int r = 0; r < 8; ++r) {
        p0s[r] = __shfl_xor(p0[r], 16, 32);
        p1s[r] = __shfl_xor(p1[r], 16, 32);
      }
      v16h bP;
#pragma unroll
      for (int j = 0; j < 8; ++j) {
        float e0, e1;
        if (j < 4) {   // lo lanes: k=2j (own st0); hi lanes: k=16+2j (partner st1)
          e0 = hi ? p1s[2 * j]     : p0[2 * j];
          e1 = hi ? p1s[2 * j + 1] : p0[2 * j + 1];
        } else {       // lo: k=2j in 8..15 (partner st0); hi: k=24.. (own st1)
          e0 = hi ? p1[2 * j - 8]  : p0s[2 * j - 8];
          e1 = hi ? p1[2 * j - 7]  : p0s[2 * j - 7];
        }
        bP[2 * j]     = (f16)e0;
        bP[2 * j + 1] = (f16)e1;
      }

      // O^T += V^T P^T : four 16x16 C frags over d (contraction = 32 keys).
#pragma unroll
      for (int db = 0; db < 4; ++db) {
        const f16* p = Vt + (db * 16 + l15) * KCHUNK + ks + hi * 16;
        union { v16h h; v8h q[2]; } a;
        a.q[0] = *(const v8h*)(p);
        a.q[1] = *(const v8h*)(p + 8);
        v8f& oo = db == 0 ? o0 : db == 1 ? o1 : db == 2 ? o2 : o3;
        oo = WMMA_F16(a.h, bP, oo);
      }
    }

    // Publish the prefetched buffer / retire the current one.
    if (wv == 0 && more) __builtin_amdgcn_s_wait_tensorcnt(0);
    __syncthreads();
  }

  // Epilogue: out = O^T / (l * temperature). Rows d consecutive per lane.
  const float inv = 1.0f / (l * 8.0f);
  float* op = out + go + (long)(qbase + l15) * HD;
#pragma unroll
  for (int db = 0; db < 4; ++db) {
    const v8f& oo = db == 0 ? o0 : db == 1 ? o1 : db == 2 ? o2 : o3;
    v4f a, b;
#pragma unroll
    for (int r = 0; r < 4; ++r) { a[r] = oo[r] * inv; b[r] = oo[r + 4] * inv; }
    *(v4f*)(op + db * 16 + hi * 8) = a;
    *(v4f*)(op + db * 16 + hi * 8 + 4) = b;
  }
}

// ---------------------------------------------------------------------------
extern "C" void kernel_launch(void* const* d_in, const int* in_sizes, int n_in,
                              void* d_out, int out_size, void* d_ws, size_t ws_size,
                              hipStream_t stream) {
  (void)in_sizes; (void)n_in; (void)out_size; (void)ws_size;
  const float* x  = (const float*)d_in[0];
  const float* Wq = (const float*)d_in[1];
  const float* bq = (const float*)d_in[2];
  const float* Wk = (const float*)d_in[3];
  const float* bk = (const float*)d_in[4];
  const float* Wv = (const float*)d_in[5];
  const float* bv = (const float*)d_in[6];
  float* out = (float*)d_out;

  f16* Qh  = (f16*)d_ws;                                  // 16 MB
  f16* Kh  = Qh + (size_t)HEADS_TOTAL * HEAD_ELEMS;       // 16 MB
  f16* VTh = Kh + (size_t)HEADS_TOTAL * HEAD_ELEMS;       // 16 MB

  // 131072 rows / 16 per wave / 8 waves per block = 1024 blocks
  qkv_proj_kernel<<<1024, 256, 0, stream>>>(x, Wq, bq, Wk, bk, Wv, bv, Qh, Kh, VTh);
  // 64 heads * 16 query tiles = 1024 blocks; 32 KB dynamic LDS (double buffer)
  attn_kernel<<<1024, 256, 2 * LDS_BUF_HALFS * sizeof(f16), stream>>>(Qh, Kh, VTh, out);
}
